// PhiMoESparseMoeBlock_18820546691499
// MI455X (gfx1250) — compile-verified, matrix operand order checked by reading, not run on hardware
//
#include <hip/hip_runtime.h>
#include <hip/hip_bf16.h>

#define NUM_EXPERTS 8
#define TOP_K 2
#define HIDDEN 2048
#define INTER 4096
#define JIT2 0.02f      // 2 * router_jitter_noise
#define LDS_STRIDE 40   // elements; 80B pitch -> conflict-free 16-lane frag reads, 16B aligned

typedef __attribute__((ext_vector_type(16))) __bf16 v16bf;
typedef __attribute__((ext_vector_type(8)))  float  v8f;

// ---------- helpers ----------

__device__ __forceinline__ unsigned short f2bfu(float f) {
    unsigned u = __float_as_uint(f);
    unsigned r = u + 0x7fffu + ((u >> 16) & 1u);   // round-to-nearest-even
    return (unsigned short)(r >> 16);
}

// 32B fragment load via two 16B-aligned pieces (LDS)
__device__ __forceinline__ v16bf ld32(const unsigned short* p) {
    struct U { uint4 a, b; };
    union { U u; v16bf v; } c;
    c.u.a = *(const uint4*)(const void*)p;
    c.u.b = *(const uint4*)(const void*)(p + 8);
    return c.v;
}

__device__ __forceinline__ v8f wmma_bf16(v16bf a, v16bf b, v8f c) {
    return __builtin_amdgcn_wmma_f32_16x16x32_bf16(
        false, a, false, b, (short)0, c, false, false);
}

// flat shared pointer -> 32-bit LDS byte address (aperture truncation rule)
__device__ __forceinline__ unsigned lds_addr(const void* p) {
    return (unsigned)(unsigned long long)p;
}

// async global->LDS copy of 16B per lane (ASYNCcnt-tracked, bypasses VGPRs)
__device__ __forceinline__ void async_b128(unsigned lds, const void* g) {
    asm volatile("global_load_async_to_lds_b128 %0, %1, off"
                 :: "v"(lds), "v"(g) : "memory");
}

__device__ __forceinline__ void wait_async0() {
    asm volatile("s_wait_asynccnt 0" ::: "memory");
}

// ---------- kernel 1: fp32 -> bf16 bulk convert ----------

__global__ void k_cvt(const float* __restrict__ x, unsigned short* __restrict__ xb, long n) {
    long i = ((long)blockIdx.x * blockDim.x + threadIdx.x) * 4;
    if (i >= n) return;
    float4 v = *(const float4*)(x + i);
    ushort4 o;
    o.x = f2bfu(v.x); o.y = f2bfu(v.y); o.z = f2bfu(v.z); o.w = f2bfu(v.w);
    *(ushort4*)(xb + i) = o;
}

// ---------- kernel 2: router ----------

__global__ void k_router(const float* __restrict__ x, const float* __restrict__ gw,
                         int* __restrict__ cnt, int* __restrict__ toklist,
                         float* __restrict__ wlist, int T) {
    int t = blockIdx.x * blockDim.x + threadIdx.x;
    if (t >= T) return;
    const float4* xr = (const float4*)(x + (size_t)t * HIDDEN);
    float acc[NUM_EXPERTS];
#pragma unroll
    for (int e = 0; e < NUM_EXPERTS; ++e) acc[e] = 0.f;
    for (int h = 0; h < HIDDEN / 4; ++h) {
        float4 xv = xr[h];
#pragma unroll
        for (int e = 0; e < NUM_EXPERTS; ++e) {
            float4 wv = ((const float4*)(gw + (size_t)e * HIDDEN))[h];
            acc[e] += xv.x * wv.x + xv.y * wv.y + xv.z * wv.z + xv.w * wv.w;
        }
    }
    int i0 = 0;
#pragma unroll
    for (int e = 1; e < NUM_EXPERTS; ++e) if (acc[e] > acc[i0]) i0 = e;
    int i1 = (i0 == 0) ? 1 : 0;
#pragma unroll
    for (int e = 0; e < NUM_EXPERTS; ++e) {
        if (e == i0) continue;
        if (acc[e] > acc[i1]) i1 = e;
    }
    float t0 = acc[i0], t1 = acc[i1];

    const float NEG = -3.0e38f;
    float m1, m2;
    {
        float s[NUM_EXPERTS], mx = NEG;
#pragma unroll
        for (int e = 0; e < NUM_EXPERTS; ++e) {
            float factor = fmaxf(fabsf(acc[e]), t0);
            bool msk = (t0 - acc[e]) / factor > JIT2;
            s[e] = msk ? NEG : acc[e];
            mx = fmaxf(mx, s[e]);
        }
        float sum = 0.f;
#pragma unroll
        for (int e = 0; e < NUM_EXPERTS; ++e)
            if (s[e] > NEG) sum += __expf(s[e] - mx);
        m1 = __expf(t0 - mx) / sum;
    }
    {
        float s[NUM_EXPERTS], mx = NEG;
#pragma unroll
        for (int e = 0; e < NUM_EXPERTS; ++e) {
            float factor = fmaxf(fabsf(acc[e]), t1);
            bool msk = ((t1 - acc[e]) / factor > JIT2) || (e == i0);
            s[e] = msk ? NEG : acc[e];
            mx = fmaxf(mx, s[e]);
        }
        float sum = 0.f;
#pragma unroll
        for (int e = 0; e < NUM_EXPERTS; ++e)
            if (s[e] > NEG) sum += __expf(s[e] - mx);
        m2 = __expf(t1 - mx) / sum;
    }
    int p0 = atomicAdd(&cnt[i0], 1);
    toklist[i0 * T + p0] = t;  wlist[i0 * T + p0] = m1;
    int p1 = atomicAdd(&cnt[i1], 1);
    toklist[i1 * T + p1] = t;  wlist[i1 * T + p1] = m2;
}

// ---------- kernel 3: tiny exclusive scan ----------

__global__ void k_scan(const int* __restrict__ cnt, int* __restrict__ basep) {
    if (threadIdx.x == 0) {
        int s = 0;
#pragma unroll
        for (int e = 0; e < NUM_EXPERTS; ++e) { basep[e] = s; s += cnt[e]; }
    }
}

// ---------- kernel 4: h = silu(x*w1^T) * (x*w3^T) ----------
// block 256 thr = 8 wave32, wave grid 2(M) x 4(N), wave tile 32x32, block tile 64x128
// double-buffered LDS fed by async global->LDS copies

__global__ void k_gemm1(const unsigned short* __restrict__ xb,
                        const unsigned short* __restrict__ w1b,
                        const unsigned short* __restrict__ w3b,
                        const int* __restrict__ cnt, const int* __restrict__ basep,
                        const int* __restrict__ toklist,
                        unsigned short* __restrict__ hbuf, int T) {
    __shared__ unsigned short sA [2][64  * LDS_STRIDE];
    __shared__ unsigned short sB1[2][128 * LDS_STRIDE];
    __shared__ unsigned short sB3[2][128 * LDS_STRIDE];
    __shared__ int sTok[64];

    const int e = blockIdx.z;
    const int count = cnt[e];
    const int mBase = blockIdx.y * 64;
    if (mBase >= count) return;                 // block-uniform exit: EXEC stays all-1
    const int nBase = blockIdx.x * 128;
    const int tid  = threadIdx.x;
    const int wave = tid >> 5;
    const int lane = tid & 31;
    const int m0l = (wave >> 2) * 32;
    const int n0l = (wave & 3) * 32;
    const int koff = (lane >> 4) * 16;
    const int lan15 = lane & 15;

    if (tid < 64) {
        int r = mBase + tid;
        sTok[tid] = toklist[e * T + ((r < count) ? r : 0)];
    }
    __syncthreads();

    const int ar = tid >> 2, ap = tid & 3;      // A: 64 rows x 4 x 16B
    const int br = tid >> 1, bp = tid & 1;      // B: 128 rows x 2 x 32B

    const unsigned short* aPtr  = xb + (size_t)sTok[ar] * HIDDEN + ap * 8;
    const unsigned short* b1Ptr = w1b + (size_t)e * INTER * HIDDEN + (size_t)(nBase + br) * HIDDEN + bp * 16;
    const unsigned short* b3Ptr = w3b + (size_t)e * INTER * HIDDEN + (size_t)(nBase + br) * HIDDEN + bp * 16;

    unsigned aDst[2], b1Dst[2], b3Dst[2];
#pragma unroll
    for (int u = 0; u < 2; ++u) {
        aDst[u]  = lds_addr(&sA [u][ar * LDS_STRIDE + ap * 8]);
        b1Dst[u] = lds_addr(&sB1[u][br * LDS_STRIDE + bp * 16]);
        b3Dst[u] = lds_addr(&sB3[u][br * LDS_STRIDE + bp * 16]);
    }

    v8f acc1[2][2], acc3[2][2];
#pragma unroll
    for (int i = 0; i < 2; ++i)
#pragma unroll
        for (int j = 0; j < 2; ++j) { acc1[i][j] = (v8f)0.f; acc3[i][j] = (v8f)0.f; }

    // prologue: batch 0 -> buffer 0
    async_b128(aDst[0],       aPtr);
    async_b128(b1Dst[0],      b1Ptr);
    async_b128(b1Dst[0] + 16, b1Ptr + 8);
    async_b128(b3Dst[0],      b3Ptr);
    async_b128(b3Dst[0] + 16, b3Ptr + 8);
    aPtr += 32; b1Ptr += 32; b3Ptr += 32;
    wait_async0();
    __syncthreads();

    for (int k = 0; k < HIDDEN; k += 32) {
        const int cur = (k >> 5) & 1;
        if (k + 32 < HIDDEN) {                  // issue next slice into other buffer
            const int nxt = cur ^ 1;
            async_b128(aDst[nxt],       aPtr);
            async_b128(b1Dst[nxt],      b1Ptr);
            async_b128(b1Dst[nxt] + 16, b1Ptr + 8);
            async_b128(b3Dst[nxt],      b3Ptr);
            async_b128(b3Dst[nxt] + 16, b3Ptr + 8);
            aPtr += 32; b1Ptr += 32; b3Ptr += 32;
        }

        v16bf a[2], b1[2], b3[2];
#pragma unroll
        for (int s = 0; s < 2; ++s) {
            a[s]  = ld32(&sA [cur][(m0l + s * 16 + lan15) * LDS_STRIDE + koff]);
            b1[s] = ld32(&sB1[cur][(n0l + s * 16 + lan15) * LDS_STRIDE + koff]);
            b3[s] = ld32(&sB3[cur][(n0l + s * 16 + lan15) * LDS_STRIDE + koff]);
        }
#pragma unroll
        for (int i = 0; i < 2; ++i)
#pragma unroll
            for (int j = 0; j < 2; ++j) {
                acc1[i][j] = wmma_bf16(a[i], b1[j], acc1[i][j]);
                acc3[i][j] = wmma_bf16(a[i], b3[j], acc3[i][j]);
            }

        wait_async0();                          // next batch landed
        __syncthreads();                        // everyone done reading cur
    }

    const size_t hb = (size_t)basep[e];
    const int rAdd = (lane >> 4) * 8;           // C/D layout: lanes 16-31 hold M=r+8
#pragma unroll
    for (int i = 0; i < 2; ++i)
#pragma unroll
        for (int j = 0; j < 2; ++j)
#pragma unroll
            for (int r = 0; r < 8; ++r) {
                int row = mBase + m0l + i * 16 + r + rAdd;
                if (row >= count) continue;
                int col = nBase + n0l + j * 16 + lan15;
                float aa = acc1[i][j][r];
                float hv = (aa / (1.f + __expf(-aa))) * acc3[i][j][r];
                hbuf[(hb + row) * (size_t)INTER + col] = f2bfu(hv);
            }
}

// ---------- kernel 5: out[token] += weight * (h * w2^T) ----------

__global__ void k_gemm2(const unsigned short* __restrict__ hbuf,
                        const unsigned short* __restrict__ w2b,
                        const int* __restrict__ cnt, const int* __restrict__ basep,
                        const int* __restrict__ toklist, const float* __restrict__ wlist,
                        float* __restrict__ out, int T) {
    __shared__ unsigned short sA[2][64  * LDS_STRIDE];
    __shared__ unsigned short sB[2][128 * LDS_STRIDE];

    const int e = blockIdx.z;
    const int count = cnt[e];
    const int mBase = blockIdx.y * 64;
    if (mBase >= count) return;
    const int nBase = blockIdx.x * 128;
    const int tid  = threadIdx.x;
    const int wave = tid >> 5;
    const int lane = tid & 31;
    const int m0l = (wave >> 2) * 32;
    const int n0l = (wave & 3) * 32;
    const int koff = (lane >> 4) * 16;
    const int lan15 = lane & 15;
    const size_t hb = (size_t)basep[e];

    const int ar = tid >> 2, ap = tid & 3;
    const int br = tid >> 1, bp = tid & 1;
    const int arow = mBase + ar;
    const unsigned short* aPtr = hbuf + (hb + ((arow < count) ? arow : 0)) * (size_t)INTER + ap * 8;
    const unsigned short* bPtr = w2b + (size_t)e * HIDDEN * INTER + (size_t)(nBase + br) * INTER + bp * 16;

    unsigned aDst[2], bDst[2];
#pragma unroll
    for (int u = 0; u < 2; ++u) {
        aDst[u] = lds_addr(&sA[u][ar * LDS_STRIDE + ap * 8]);
        bDst[u] = lds_addr(&sB[u][br * LDS_STRIDE + bp * 16]);
    }

    v8f acc[2][2];
#pragma unroll
    for (int i = 0; i < 2; ++i)
#pragma unroll
        for (int j = 0; j < 2; ++j) acc[i][j] = (v8f)0.f;

    async_b128(aDst[0],      aPtr);
    async_b128(bDst[0],      bPtr);
    async_b128(bDst[0] + 16, bPtr + 8);
    aPtr += 32; bPtr += 32;
    wait_async0();
    __syncthreads();

    for (int k = 0; k < INTER; k += 32) {
        const int cur = (k >> 5) & 1;
        if (k + 32 < INTER) {
            const int nxt = cur ^ 1;
            async_b128(aDst[nxt],      aPtr);
            async_b128(bDst[nxt],      bPtr);
            async_b128(bDst[nxt] + 16, bPtr + 8);
            aPtr += 32; bPtr += 32;
        }

        v16bf a[2], b[2];
#pragma unroll
        for (int s = 0; s < 2; ++s) {
            a[s] = ld32(&sA[cur][(m0l + s * 16 + lan15) * LDS_STRIDE + koff]);
            b[s] = ld32(&sB[cur][(n0l + s * 16 + lan15) * LDS_STRIDE + koff]);
        }
#pragma unroll
        for (int i = 0; i < 2; ++i)
#pragma unroll
            for (int j = 0; j < 2; ++j)
                acc[i][j] = wmma_bf16(a[i], b[j], acc[i][j]);

        wait_async0();
        __syncthreads();
    }

    const int rAdd = (lane >> 4) * 8;
#pragma unroll
    for (int i = 0; i < 2; ++i)
#pragma unroll
        for (int j = 0; j < 2; ++j)
#pragma unroll
            for (int r = 0; r < 8; ++r) {
                int row = mBase + m0l + i * 16 + r + rAdd;
                if (row >= count) continue;
                int tok = toklist[e * T + row];
                float wt = wlist[e * T + row];
                int col = nBase + n0l + j * 16 + lan15;
                atomicAdd(&out[(size_t)tok * HIDDEN + col], acc[i][j][r] * wt);
            }
}

// ---------- launch ----------

extern "C" void kernel_launch(void* const* d_in, const int* in_sizes, int n_in,
                              void* d_out, int out_size, void* d_ws, size_t ws_size,
                              hipStream_t stream) {
    const float* x      = (const float*)d_in[0];   // [2,2048,2048]
    const float* gate_w = (const float*)d_in[1];   // [8,2048]
    const float* w1     = (const float*)d_in[2];   // [8,4096,2048]
    const float* w2     = (const float*)d_in[3];   // [8,2048,4096]
    const float* w3     = (const float*)d_in[4];   // [8,4096,2048]

    const int  T  = in_sizes[0] / HIDDEN;          // 4096 tokens
    const long nX = (long)in_sizes[0];
    const long nW = (long)NUM_EXPERTS * INTER * HIDDEN;

    char* ws = (char*)d_ws;
    size_t off = 0;
    unsigned short* xb  = (unsigned short*)(ws + off);  off += (size_t)T * HIDDEN * 2;
    unsigned short* w1b = (unsigned short*)(ws + off);  off += (size_t)nW * 2;
    unsigned short* w3b = (unsigned short*)(ws + off);  off += (size_t)nW * 2;
    unsigned short* w2b = (unsigned short*)(ws + off);  off += (size_t)nW * 2;
    int*   cnt     = (int*)(ws + off);                  off += 256;
    int*   basep   = (int*)(ws + off);                  off += 256;
    int*   toklist = (int*)(ws + off);                  off += (size_t)NUM_EXPERTS * T * 4;
    float* wlist   = (float*)(ws + off);                off += (size_t)NUM_EXPERTS * T * 4;
    unsigned short* hbuf = (unsigned short*)(ws + off); // 2T * I * 2 = 64 MB

    hipMemsetAsync(cnt, 0, NUM_EXPERTS * sizeof(int), stream);
    hipMemsetAsync(d_out, 0, (size_t)out_size * sizeof(float), stream);

    const int blk = 256;
    k_cvt<<<(unsigned)((nX / 4 + blk - 1) / blk), blk, 0, stream>>>(x, xb, nX);
    k_cvt<<<(unsigned)((nW / 4 + blk - 1) / blk), blk, 0, stream>>>(w1, w1b, nW);
    k_cvt<<<(unsigned)((nW / 4 + blk - 1) / blk), blk, 0, stream>>>(w3, w3b, nW);
    k_cvt<<<(unsigned)((nW / 4 + blk - 1) / blk), blk, 0, stream>>>(w2, w2b, nW);

    k_router<<<(T + blk - 1) / blk, blk, 0, stream>>>(x, gate_w, cnt, toklist, wlist, T);
    k_scan<<<1, 32, 0, stream>>>(cnt, basep);

    {
        dim3 g(INTER / 128, (T + 63) / 64, NUM_EXPERTS);
        k_gemm1<<<g, 256, 0, stream>>>(xb, w1b, w3b, cnt, basep, toklist, hbuf, T);
    }
    {
        dim3 g(HIDDEN / 128, (T + 63) / 64, NUM_EXPERTS);
        k_gemm2<<<g, 256, 0, stream>>>(hbuf, w2b, cnt, basep, toklist, wlist, (float*)d_out, T);
    }
}
